// MultiScaleRetention_34153579938379
// MI455X (gfx1250) — matrix operand order repaired and checked by reading, not compiled, not run on hardware
//
#include <hip/hip_runtime.h>
#include <hip/hip_bf16.h>

// MultiScaleRetention forward for MI455X (gfx1250).
// bf16 WMMA GEMMs fed by async global->LDS copies (2-stage software pipeline,
// compile-time buffer addresses, K-step 64), chunkwise-retention scan as WMMA
// GEMMs, wave-level groupnorm+gate.
// B=2 N=2048 H=16 DK=DV=128 E=2048.  d_ws usage ~184 MB.

#define B_ 2
#define N_ 2048
#define H_ 16
#define DK_ 128
#define DV_ 128
#define E_ 2048
#define MM_ (B_ * N_)   // 4096 rows
#define CHUNK 64

typedef __attribute__((ext_vector_type(16))) __bf16 bf16x16;
typedef __attribute__((ext_vector_type(8)))  __bf16 bf16x8;
typedef __attribute__((ext_vector_type(4)))  __bf16 bf16x4;
typedef __attribute__((ext_vector_type(8)))  float  f32x8;

__device__ __forceinline__ __bf16 bf(float f) {
    unsigned int u = __float_as_uint(f);
    u += 0x7FFFu + ((u >> 16) & 1u);          // round-to-nearest-even bf16
    unsigned short s = (unsigned short)(u >> 16);
    __bf16 b;
    __builtin_memcpy(&b, &s, 2);
    return b;
}

// A-matrix fragment (16x32 bf16): lane holds row M=lane%16,
// K = (lane/16)*16 + [0..15] contiguous -> two 16B LDS loads.
__device__ __forceinline__ bf16x16 fragA(const __bf16* rowbase, int kk, int lane) {
    const __bf16* p = rowbase + kk + ((lane >> 4) << 4);
    bf16x8 lo = *(const bf16x8*)p;
    bf16x8 hi = *(const bf16x8*)(p + 8);
    return __builtin_shufflevector(lo, hi, 0,1,2,3,4,5,6,7,8,9,10,11,12,13,14,15);
}

// B-matrix fragment (32x16 bf16): lane holds col N=lane%16,
// K = (lane/16)*16 + [0..15] contiguous -> two 16B LDS loads.
__device__ __forceinline__ bf16x16 fragB(const __bf16* rowbase, int kk, int lane) {
    const __bf16* p = rowbase + kk + ((lane >> 4) << 4);
    bf16x8 lo = *(const bf16x8*)p;
    bf16x8 hi = *(const bf16x8*)(p + 8);
    return __builtin_shufflevector(lo, hi, 0,1,2,3,4,5,6,7,8,9,10,11,12,13,14,15);
}

// fragB from f32 LDS (retention state), converted in-regs.
__device__ __forceinline__ bf16x16 fragB_f32(const float* rowbase, int kk, int lane) {
    const float* p = rowbase + kk + ((lane >> 4) << 4);
    bf16x16 o;
#pragma unroll
    for (int i = 0; i < 16; ++i) o[i] = bf(p[i]);
    return o;
}

__device__ __forceinline__ f32x8 wmma_bf16(bf16x16 a, bf16x16 b, f32x8 c) {
    return __builtin_amdgcn_wmma_f32_16x16x32_bf16(false, a, false, b, (short)0, c,
                                                   false, false);
}

// Async copy of 16 bytes from global (saddr64 + 32-bit byte offset) to LDS.
__device__ __forceinline__ void async_b128(unsigned lds_addr, unsigned goff,
                                           unsigned long long gbase) {
    asm volatile("global_load_async_to_lds_b128 %0, %1, %2 offset:0"
                 :: "v"(lds_addr), "v"(goff), "s"(gbase) : "memory");
}

// ---------------------------------------------------------------------------
// f32 -> bf16 bulk convert (8 elements / thread).
// ---------------------------------------------------------------------------
__global__ __launch_bounds__(256) void f32_to_bf16_kernel(
    const float* __restrict__ in, __bf16* __restrict__ out, int n8)
{
    const int i = blockIdx.x * 256 + threadIdx.x;
    if (i >= n8) return;
    const float4* p = (const float4*)in + (size_t)i * 2;
    float4 a = p[0], b = p[1];
    bf16x8 o;
    o[0] = bf(a.x); o[1] = bf(a.y); o[2] = bf(a.z); o[3] = bf(a.w);
    o[4] = bf(b.x); o[5] = bf(b.y); o[6] = bf(b.z); o[7] = bf(b.w);
    ((bf16x8*)out)[i] = o;
}

// ---------------------------------------------------------------------------
// C[4096,2048] = epilogue( A[4096,2048] @ W[2048,2048]^T + bias ), bf16 A/W.
// 2-stage async pipeline, K-step 64, 8 waves, 32x64 tile per wave.
// MODE 0: f32 out    MODE 1: bf16 out * scale
// MODE 2: f32 silu   MODE 3: bf16 out
// ---------------------------------------------------------------------------
template <int MODE>
__global__ __launch_bounds__(256) void gemm_bf16_async_kernel(
    const __bf16* __restrict__ A, const __bf16* __restrict__ W,
    const float* __restrict__ bias, void* __restrict__ Cout, float scale)
{
    constexpr int K = E_;
    constexpr int N = E_;
    constexpr unsigned RS = 144u;          // LDS row stride bytes (72 bf16)

    __shared__ __attribute__((aligned(16))) __bf16 sA[2][128][72];
    __shared__ __attribute__((aligned(16))) __bf16 sW[2][128][72];

    const int tid  = threadIdx.x;
    const int bm   = blockIdx.y * 128;
    const int bn   = blockIdx.x * 128;
    const int w    = tid >> 5;
    const int lane = tid & 31;
    const int wm   = (w & 3) * 32;
    const int wn   = (w >> 2) * 64;
    const int mrow = lane & 15;
    const int mhi  = (lane >> 4) * 8;

    // Staging: 128 rows x 64 bf16 = 1024 x 16B chunks; thread owns 4 rows
    // (r0 + 32j), 16B segment s0 within each row.
    const int r0 = tid >> 3;               // 0..31
    const int s0 = (tid & 7) * 8;          // 0..56

    const unsigned sAb[2] = { (unsigned)(uintptr_t)&sA[0][0][0],
                              (unsigned)(uintptr_t)&sA[1][0][0] };
    const unsigned sWb[2] = { (unsigned)(uintptr_t)&sW[0][0][0],
                              (unsigned)(uintptr_t)&sW[1][0][0] };
    const unsigned long long Ab = (unsigned long long)(uintptr_t)A;
    const unsigned long long Wb = (unsigned long long)(uintptr_t)W;
    const unsigned lo = (unsigned)r0 * RS + (unsigned)s0 * 2u;

    f32x8 acc[2][4] = {};

    // issue one 128x64 A-tile + W-tile into buffer `buf` for k-offset k0
    auto issue = [&](int bufc, int k0) {
#pragma unroll
        for (int j = 0; j < 4; ++j) {
            const unsigned l = lo + (unsigned)(32 * j) * RS;
            async_b128(sAb[bufc] + l, (((unsigned)(bm + r0 + 32 * j) * K) + k0 + s0) * 2u, Ab);
            async_b128(sWb[bufc] + l, (((unsigned)(bn + r0 + 32 * j) * K) + k0 + s0) * 2u, Wb);
        }
    };
    auto compute = [&](int bufc) {
#pragma unroll
        for (int kk = 0; kk < 64; kk += 32) {
            bf16x16 af[2], bfw[4];
#pragma unroll
            for (int mi = 0; mi < 2; ++mi)
                af[mi] = fragA(&sA[bufc][wm + mi * 16 + mrow][0], kk, lane);
#pragma unroll
            for (int ni = 0; ni < 4; ++ni)
                bfw[ni] = fragB(&sW[bufc][wn + ni * 16 + mrow][0], kk, lane);
#pragma unroll
            for (int mi = 0; mi < 2; ++mi)
#pragma unroll
                for (int ni = 0; ni < 4; ++ni)
                    acc[mi][ni] = wmma_bf16(af[mi], bfw[ni], acc[mi][ni]);
        }
    };

    issue(0, 0);
    for (int k0 = 0; k0 < K; k0 += 128) {       // two pipeline stages / iter
        // ---- stage A: data for k0 in buf0; prefetch k0+64 -> buf1 ----
        if (k0 + 64 < K) {
            issue(1, k0 + 64);
            asm volatile("s_wait_asynccnt 0x8" ::: "memory");
        } else {
            asm volatile("s_wait_asynccnt 0x0" ::: "memory");
        }
        __syncthreads();
        compute(0);
        __syncthreads();
        // ---- stage B: data for k0+64 in buf1; prefetch k0+128 -> buf0 ----
        if (k0 + 128 < K) {
            issue(0, k0 + 128);
            asm volatile("s_wait_asynccnt 0x8" ::: "memory");
        } else {
            asm volatile("s_wait_asynccnt 0x0" ::: "memory");
        }
        __syncthreads();
        compute(1);
        __syncthreads();
    }

#pragma unroll
    for (int mi = 0; mi < 2; ++mi)
#pragma unroll
        for (int ni = 0; ni < 4; ++ni)
#pragma unroll
            for (int vv = 0; vv < 8; ++vv) {
                const int row = bm + wm + mi * 16 + mhi + vv;
                const int col = bn + wn + ni * 16 + mrow;
                float val = acc[mi][ni][vv] + bias[col];
                const size_t idx = (size_t)row * N + col;
                if (MODE == 0) {
                    ((float*)Cout)[idx] = val;
                } else if (MODE == 1) {
                    ((__bf16*)Cout)[idx] = bf(val * scale);
                } else if (MODE == 2) {
                    ((float*)Cout)[idx] = val * (1.f / (1.f + __expf(-val)));
                } else {
                    ((__bf16*)Cout)[idx] = bf(val);
                }
            }
}

// ---------------------------------------------------------------------------
// Chunkwise retention scan (q,k,v are bf16). One workgroup per (b,h).
//   Kd[s]   = K[s] * l^(C-1-s)
//   Am[i,j] = (i>=j) ? l^(i+1-C) * (Q Kd^T)[i,j] : 0
//   ret[i]  = (Am V)[i] + l^(i+1) * (Q S)[i]
//   S_t     = l^C * S_t + V^T Kd          (S_t[v][d] == state[d][v])
// ---------------------------------------------------------------------------
__global__ __launch_bounds__(256) void retention_scan_kernel(
    const __bf16* __restrict__ q, const __bf16* __restrict__ k,
    const __bf16* __restrict__ v, float* __restrict__ ret,
    float* __restrict__ state_out)
{
    __shared__ __attribute__((aligned(16))) __bf16 sQ  [CHUNK][136];  // [i][d]
    __shared__ __attribute__((aligned(16))) __bf16 sKd [CHUNK][136];  // [j][d]
    __shared__ __attribute__((aligned(16))) __bf16 sKdT[DK_ ][72];    // [d][s]
    __shared__ __attribute__((aligned(16))) __bf16 sVt [DV_ ][72];    // [v][s]
    __shared__ __attribute__((aligned(16))) __bf16 sAm [CHUNK][72];   // [i][j]
    __shared__ __attribute__((aligned(16))) float  sS  [DV_ ][136];   // [v][d]

    const int tid  = threadIdx.x;
    const int bh   = blockIdx.x;               // 0..31
    const int b    = bh / H_;
    const int h    = bh % H_;
    const float lambda = 1.f - exp2f(-5.f - (float)h);
    const float l2     = log2f(lambda);

    const int w    = tid >> 5;
    const int lane = tid & 31;
    const int mrow = lane & 15;
    const int mhi  = (lane >> 4) * 8;

    for (int i = tid; i < DV_ * 136; i += 256) ((float*)sS)[i] = 0.f;

    const int lr  = tid >> 2;            // 0..63 (chunk row)
    const int seg = (tid & 3) * 32;      // feature segment

    for (int ch = 0; ch < N_ / CHUNK; ++ch) {
        const int n0 = ch * CHUNK;
        __syncthreads();
        {   // --- stage Q, decayed K (both layouts), V^T into LDS ---
            const size_t base = (((size_t)b * N_ + n0 + lr) * H_ + h) * (size_t)DK_;
            const float ksc = exp2f(l2 * (float)(CHUNK - 1 - lr));
#pragma unroll
            for (int u = 0; u < 4; ++u) {
                const int d = seg + u * 8;
                bf16x8 qv = *(const bf16x8*)&q[base + d];
                *(bf16x8*)&sQ[lr][d] = qv;
                bf16x8 kv = *(const bf16x8*)&k[base + d];
                bf16x8 vv = *(const bf16x8*)&v[base + d];
#pragma unroll
                for (int e = 0; e < 8; ++e) {
                    __bf16 kb = bf((float)kv[e] * ksc);
                    sKd[lr][d + e]  = kb;
                    sKdT[d + e][lr] = kb;
                    sVt[d + e][lr]  = vv[e];
                }
            }
        }
        __syncthreads();

        {   // --- GEMM1: Am = causal-mask( Q @ Kd^T ), 64x64, K=128 ---
            const int tm  = w & 3;
            const int tnb = (w >> 2) * 2;
            f32x8 a1[2] = {};
            for (int kk = 0; kk < DK_; kk += 32) {
                bf16x16 aq = fragA(&sQ[tm * 16 + mrow][0], kk, lane);
#pragma unroll
                for (int ni = 0; ni < 2; ++ni) {
                    bf16x16 bk = fragB(&sKd[(tnb + ni) * 16 + mrow][0], kk, lane);
                    a1[ni] = wmma_bf16(aq, bk, a1[ni]);
                }
            }
#pragma unroll
            for (int ni = 0; ni < 2; ++ni)
#pragma unroll
                for (int vv = 0; vv < 8; ++vv) {
                    const int i = tm * 16 + mhi + vv;
                    const int j = (tnb + ni) * 16 + mrow;
                    float val = (i >= j)
                        ? exp2f(l2 * (float)(i + 1 - CHUNK)) * a1[ni][vv] : 0.f;
                    sAm[i][j] = bf(val);
                }
        }
        __syncthreads();

        {   // --- ret = Am @ V  +  l^(i+1) * (Q @ S), 64x128 ---
            const int tm  = w & 3;
            const int tnb = (w >> 2) * 4;
            f32x8 ain[4] = {}, acr[4] = {};
            for (int kk = 0; kk < CHUNK; kk += 32) {
                bf16x16 aa = fragA(&sAm[tm * 16 + mrow][0], kk, lane);
#pragma unroll
                for (int ni = 0; ni < 4; ++ni) {
                    bf16x16 bb = fragB(&sVt[(tnb + ni) * 16 + mrow][0], kk, lane);
                    ain[ni] = wmma_bf16(aa, bb, ain[ni]);
                }
            }
            for (int kk = 0; kk < DK_; kk += 32) {
                bf16x16 aq = fragA(&sQ[tm * 16 + mrow][0], kk, lane);
#pragma unroll
                for (int ni = 0; ni < 4; ++ni) {
                    bf16x16 bs = fragB_f32(&sS[(tnb + ni) * 16 + mrow][0], kk, lane);
                    acr[ni] = wmma_bf16(aq, bs, acr[ni]);
                }
            }
#pragma unroll
            for (int ni = 0; ni < 4; ++ni)
#pragma unroll
                for (int vv = 0; vv < 8; ++vv) {
                    const int i  = tm * 16 + mhi + vv;
                    const int vc = (tnb + ni) * 16 + mrow;
                    float r = ain[ni][vv] + exp2f(l2 * (float)(i + 1)) * acr[ni][vv];
                    ret[(((size_t)b * N_ + n0 + i) * H_ + h) * DV_ + vc] = r;
                }
        }
        __syncthreads();

        {   // --- state update: S_t = l^C * S_t + V^T @ Kd, 128x128, K=64 ---
            const int tm = w;
            const float decC = exp2f(l2 * (float)CHUNK);
            f32x8 au[8] = {};
            for (int kk = 0; kk < CHUNK; kk += 32) {
                bf16x16 av = fragA(&sVt[tm * 16 + mrow][0], kk, lane);
#pragma unroll
                for (int ni = 0; ni < 8; ++ni) {
                    bf16x16 bkd = fragB(&sKdT[ni * 16 + mrow][0], kk, lane);
                    au[ni] = wmma_bf16(av, bkd, au[ni]);
                }
            }
#pragma unroll
            for (int ni = 0; ni < 8; ++ni)
#pragma unroll
                for (int vv = 0; vv < 8; ++vv) {
                    const int vrow = tm * 16 + mhi + vv;
                    const int dcol = ni * 16 + mrow;
                    sS[vrow][dcol] = decC * sS[vrow][dcol] + au[ni][vv];
                }
        }
    }
    __syncthreads();

    // final state: state_out[((b*H + h)*DK + d)*DV + v] = sS[v][d]
    for (int i = tid; i < DK_ * DV_; i += 256) {
        const int d  = i / DV_;
        const int vv = i % DV_;
        state_out[(((size_t)b * H_ + h) * DK_ + d) * DV_ + vv] = sS[vv][d];
    }
}

// ---------------------------------------------------------------------------
// Per-(b,n,h) layernorm over DV=128 times silu gate; bf16 output for final GEMM.
// One wave per head vector; 8 waves / block.
// ---------------------------------------------------------------------------
__global__ __launch_bounds__(256) void norm_gate_kernel(
    const float* __restrict__ ret, const float* __restrict__ gate,
    __bf16* __restrict__ y)
{
    const int idx  = blockIdx.x * 8 + (threadIdx.x >> 5);  // (b*N+n)*H + h
    const int lane = threadIdx.x & 31;
    const float* r = ret + (size_t)idx * DV_;
    float4 rv = *(const float4*)&r[lane * 4];
    float s  = rv.x + rv.y + rv.z + rv.w;
    float s2 = rv.x * rv.x + rv.y * rv.y + rv.z * rv.z + rv.w * rv.w;
#pragma unroll
    for (int off = 16; off > 0; off >>= 1) {
        s  += __shfl_xor(s,  off, 32);
        s2 += __shfl_xor(s2, off, 32);
    }
    const float mu  = s * (1.f / DV_);
    const float var = s2 * (1.f / DV_) - mu * mu;
    const float inv = rsqrtf(var + 1e-6f);
    const float* g  = gate + (size_t)idx * DV_;
    float4 gv = *(const float4*)&g[lane * 4];
    bf16x4 o;
    o[0] = bf((rv.x - mu) * inv * gv.x);
    o[1] = bf((rv.y - mu) * inv * gv.y);
    o[2] = bf((rv.z - mu) * inv * gv.z);
    o[3] = bf((rv.w - mu) * inv * gv.w);
    *(bf16x4*)&y[(size_t)idx * DV_ + lane * 4] = o;
}

extern "C" void kernel_launch(void* const* d_in, const int* in_sizes, int n_in,
                              void* d_out, int out_size, void* d_ws, size_t ws_size,
                              hipStream_t stream)
{
    const float* x  = (const float*)d_in[0];
    const float* Wq = (const float*)d_in[1];
    const float* bq = (const float*)d_in[2];
    const float* Wk = (const float*)d_in[3];
    const float* bk = (const float*)d_in[4];
    const float* Wv = (const float*)d_in[5];
    const float* bv = (const float*)d_in[6];
    const float* Wg = (const float*)d_in[7];
    const float* bg = (const float*)d_in[8];
    const float* Wo = (const float*)d_in[9];
    const float* bo = (const float*)d_in[10];
    (void)in_sizes; (void)n_in; (void)out_size; (void)ws_size;

    float* out       = (float*)d_out;
    float* state_out = out + (size_t)B_ * N_ * E_;

    const size_t SZ  = (size_t)B_ * N_ * E_;   // 8,388,608 activations
    const size_t WSZ = (size_t)E_ * E_;        // 4,194,304 per weight

    // bf16 region
    __bf16* xb  = (__bf16*)d_ws;
    __bf16* wqb = xb  + SZ;
    __bf16* wkb = wqb + WSZ;
    __bf16* wvb = wkb + WSZ;
    __bf16* wgb = wvb + WSZ;
    __bf16* wob = wgb + WSZ;
    __bf16* qb  = wob + WSZ;
    __bf16* kb  = qb  + SZ;
    __bf16* vb  = kb  + SZ;
    __bf16* ybf = vb  + SZ;
    // f32 region (16B aligned: all offsets above are multiples of 8 elements)
    float* gb = (float*)(ybf + SZ);
    float* rb = gb + SZ;

    dim3 gg(E_ / 128, MM_ / 128);              // (16, 32)
    dim3 bb(256);
    const float kscale = 1.f / sqrtf((float)DK_);

    // 1) bulk f32 -> bf16 conversions (x + 5 weights)
    f32_to_bf16_kernel<<<(int)(SZ  / 8 / 256), bb, 0, stream>>>(x,  xb,  (int)(SZ  / 8));
    f32_to_bf16_kernel<<<(int)(WSZ / 8 / 256), bb, 0, stream>>>(Wq, wqb, (int)(WSZ / 8));
    f32_to_bf16_kernel<<<(int)(WSZ / 8 / 256), bb, 0, stream>>>(Wk, wkb, (int)(WSZ / 8));
    f32_to_bf16_kernel<<<(int)(WSZ / 8 / 256), bb, 0, stream>>>(Wv, wvb, (int)(WSZ / 8));
    f32_to_bf16_kernel<<<(int)(WSZ / 8 / 256), bb, 0, stream>>>(Wg, wgb, (int)(WSZ / 8));
    f32_to_bf16_kernel<<<(int)(WSZ / 8 / 256), bb, 0, stream>>>(Wo, wob, (int)(WSZ / 8));

    // 2) projections (async-staged bf16 WMMA GEMMs)
    gemm_bf16_async_kernel<3><<<gg, bb, 0, stream>>>(xb, wqb, bq, qb, 1.f);
    gemm_bf16_async_kernel<1><<<gg, bb, 0, stream>>>(xb, wkb, bk, kb, kscale);
    gemm_bf16_async_kernel<3><<<gg, bb, 0, stream>>>(xb, wvb, bv, vb, 1.f);
    gemm_bf16_async_kernel<2><<<gg, bb, 0, stream>>>(xb, wgb, bg, gb, 1.f);

    // 3) chunkwise retention scan (writes ret f32 + final state into d_out tail)
    retention_scan_kernel<<<B_ * H_, 256, 0, stream>>>(qb, kb, vb, rb, state_out);

    // 4) groupnorm * silu gate -> bf16 y
    norm_gate_kernel<<<(B_ * N_ * H_) / 8, bb, 0, stream>>>(rb, gb, ybf);

    // 5) output projection
    gemm_bf16_async_kernel<0><<<gg, bb, 0, stream>>>(ybf, wob, bo, out, 1.f);
}
